// HRNetV2MambaBackbone_790273982896
// MI455X (gfx1250) — matrix operand order repaired and checked by reading, not compiled, not run on hardware
//
#include <hip/hip_runtime.h>
#include <hip/hip_bf16.h>

// HRNetV2 backbone inference for gfx1250 (MI455X).
//  - Conv weights converted f32->f16 once per launch into a d_ws arena.
//  - Each conv+BN(+ReLU/residual/accumulate) is one implicit-GEMM WMMA kernel:
//      v_wmma_f32_16x16x32_f16, block tile 64(M) x 64(N), 4 wave32 per block,
//      4 WMMAs per wave per K-step (B fragment reused across four M sub-tiles).
//  - Double-buffered LDS, one barrier per K-step; next step's weight tile is
//    DMA'd with global_load_async_to_lds_b128 (ASYNCcnt) while WMMAs run.
//  - Template-specialized gathers: 1x1 (s1p0, no bounds checks), 3x3 s1,
//    3x3 runtime-stride, and a ragged-K variant only for the stem (K=27).

typedef __attribute__((ext_vector_type(16))) _Float16 v16h;
typedef __attribute__((ext_vector_type(8)))  _Float16 v8h;
typedef __attribute__((ext_vector_type(8)))  float    v8f;

#define LDS_K 40   // 32 K-halves + pad (80B row stride, keeps 16B alignment)

// ---------------------------------------------------------------------------
// KH/KW: filter size (1 or 3). SR: compile-time stride (0 = runtime).
// RAG: Ktot not a multiple of 32 (stem only) -> keep kg<Ktot checks + fallback.
// pad is (KH==3) ? 1 : 0 at compile time.
// ---------------------------------------------------------------------------
template<int KH, int KW, int SR, bool RAG>
__global__ __launch_bounds__(128)
void conv_bn_wmma(const float* __restrict__ x, const _Float16* __restrict__ wgt,
                  const float* __restrict__ bng, const float* __restrict__ bnb,
                  const float* __restrict__ bnm, const float* __restrict__ bnv,
                  const float* __restrict__ res, float* __restrict__ out,
                  int NB, int Cin, int H, int W, int Cout, int OH, int OW,
                  int rstride, int relu, int accum)
{
    __shared__ _Float16 As[2][64][LDS_K];      // ping-pong [m][k]
    __shared__ _Float16 Bs[2][4][16][LDS_K];   // ping-pong per wave [n][k]

    constexpr int PAD = (KH == 3) ? 1 : 0;
    const int str = SR ? SR : rstride;

    const int tid  = threadIdx.x;
    const int wv   = tid >> 5;
    const int lane = tid & 31;
    const int half = lane >> 4;
    const int lc   = lane & 15;

    const int Mtot = NB * OH * OW;
    const int m0   = blockIdx.x * 64;
    const int n0   = (blockIdx.y * 4 + wv) * 16;
    const int Ktot = Cin * KH * KW;
    const int Ksteps = (Ktot + 31) >> 5;
    const int HWin = H * W;
    const int OHW  = OH * OW;
    const bool nvalid = (n0 < Cout);                       // wave-uniform
    const unsigned bb[2] = { (unsigned)(size_t)&Bs[0][wv][0][0],
                             (unsigned)(size_t)&Bs[1][wv][0][0] };

    // ---- per-thread A-chunk constants (2 chunks of 8 consecutive k) --------
    int mi_c[2], kb_c[2], boff_c[2], ohs_c[2], ows_c[2];
    bool mok_c[2];
    #pragma unroll
    for (int ch = 0; ch < 2; ++ch) {
        int idx = tid * 16 + ch * 8;           // 0..2047
        int mi = idx >> 5, kb = idx & 31;
        int s = m0 + mi;
        bool mo = s < Mtot;
        int ss = mo ? s : 0;
        int b  = ss / OHW;
        int sp = ss - b * OHW;
        int oh = sp / OW, ow = sp - oh * OW;
        mi_c[ch] = mi; kb_c[ch] = kb; mok_c[ch] = mo;
        boff_c[ch] = b * Cin * HWin;
        ohs_c[ch]  = oh * str - PAD;
        ows_c[ch]  = ow * str - PAD;
    }

    // ---- staging lambdas ---------------------------------------------------
    auto stageA = [&](int buf, int ks) {
        #pragma unroll
        for (int ch = 0; ch < 2; ++ch) {
            float fv[8];
            bool  okf[8];
            #pragma unroll
            for (int e = 0; e < 8; ++e) {      // batch the 8 global loads
                int kg = ks * 32 + kb_c[ch] + e;
                int ci = kg / (KH * KW);
                int rr = kg - ci * (KH * KW);
                int kh = rr / KW, kw = rr - kh * KW;
                int ih = ohs_c[ch] + kh, iw = ows_c[ch] + kw;
                bool ok;
                if constexpr (KH == 1)
                    ok = mok_c[ch];            // s1p0: always spatially in-bounds
                else
                    ok = mok_c[ch] & (ih >= 0) & (ih < H) & (iw >= 0) & (iw < W);
                if constexpr (RAG)
                    ok = ok & (kg < Ktot);
                int gix = ok ? (boff_c[ch] + ci * HWin + ih * W + iw) : 0;
                fv[e] = x[gix];                // unconditional, clamped index
                okf[e] = ok;
            }
            v8h pk;
            #pragma unroll
            for (int e = 0; e < 8; ++e)
                pk[e] = okf[e] ? (_Float16)fv[e] : (_Float16)0.f;
            *(v8h*)&As[buf][mi_c[ch]][kb_c[ch]] = pk;   // one ds_store_b128
        }
    };
    auto stageB = [&](int buf, int ks) {
        if (nvalid && (!RAG || (ks * 32 + 32 <= Ktot))) {
            // fully in-bounds: async DMA straight into LDS (ASYNCcnt path)
            #pragma unroll
            for (int e = 0; e < 2; ++e) {
                int idx = lane * 2 + e;                 // 64 x 16B chunks
                int ni  = idx >> 2;
                int kk  = (idx & 3) * 8;
                unsigned lds = bb[buf] + (unsigned)((ni * LDS_K + kk) * 2);
                const _Float16* g =
                    wgt + (size_t)(n0 + ni) * Ktot + (size_t)ks * 32 + kk;
                asm volatile("global_load_async_to_lds_b128 %0, %1, off"
                             :: "v"(lds), "v"(g) : "memory");
            }
        } else if (RAG && nvalid) {
            // ragged K edge (stem conv only)
            #pragma unroll
            for (int e = 0; e < 16; ++e) {
                int idx = lane * 16 + e;
                int ni = idx >> 5, kk = idx & 31;
                int kg = ks * 32 + kk;
                _Float16 v = (_Float16)0.f;
                if (kg < Ktot) v = wgt[(size_t)(n0 + ni) * Ktot + kg];
                Bs[buf][wv][ni][kk] = v;
            }
        }
    };

    v8f acc[4] = {{}, {}, {}, {}};

    stageA(0, 0);
    stageB(0, 0);
    for (int ks = 0; ks < Ksteps; ++ks) {
        __syncthreads();                               // cur buffer complete
        asm volatile("s_wait_asynccnt 0x0" ::: "memory");  // own B DMA landed
        int buf = ks & 1;

        v16h a[4], bv;
        #pragma unroll
        for (int p = 0; p < 8; ++p) {
            int ka = (p < 4) ? (half * 8 + 2 * p) : (16 + half * 8 + 2 * (p - 4));
            int kb = (p < 4) ? (half * 8 + 2 * p) : (16 + half * 8 + 2 * (p - 4));
            #pragma unroll
            for (int sub = 0; sub < 4; ++sub) {
                a[sub][2 * p]     = As[buf][sub * 16 + lc][ka];
                a[sub][2 * p + 1] = As[buf][sub * 16 + lc][ka + 1];
            }
            bv[2 * p]     = Bs[buf][wv][lc][kb];
            bv[2 * p + 1] = Bs[buf][wv][lc][kb + 1];
        }
        if (ks + 1 < Ksteps) {                         // overlap next staging
            stageA(buf ^ 1, ks + 1);
            stageB(buf ^ 1, ks + 1);
        }
        #pragma unroll
        for (int sub = 0; sub < 4; ++sub)
            acc[sub] = __builtin_amdgcn_wmma_f32_16x16x32_f16(
                           false, a[sub], false, bv, (short)0, acc[sub],
                           false, false);
    }

    // ---- epilogue: folded BN, residual, accumulate, ReLU -------------------
    int n = n0 + lc;
    if (n < Cout) {
        float sc = bng[n] * rsqrtf(bnv[n] + 1e-5f);
        float sh = bnb[n] - bnm[n] * sc;
        #pragma unroll
        for (int sub = 0; sub < 4; ++sub) {
            #pragma unroll
            for (int r = 0; r < 8; ++r) {
                int ml = sub * 16 + r + 8 * half;      // C/D: VGPR r -> M=r+8*half
                int s  = m0 + ml;
                if (s < Mtot) {
                    int b  = s / OHW;
                    int sp = s - b * OHW;
                    int oh = sp / OW, ow = sp - oh * OW;
                    size_t oidx = (((size_t)b * Cout + n) * OH + oh) * OW + ow;
                    float v = acc[sub][r] * sc + sh;
                    if (res)   v += res[oidx];
                    if (accum) v += out[oidx];
                    if (relu)  v = fmaxf(v, 0.f);
                    out[oidx] = v;
                }
            }
        }
    }
}

// ---------------------------------------------------------------------------
__global__ void f32_to_f16_k(const float* __restrict__ in,
                             _Float16* __restrict__ out, size_t n)
{
    size_t i = (size_t)blockIdx.x * blockDim.x + threadIdx.x;
    if (i < n) out[i] = (_Float16)in[i];
}

__global__ void upsample_bilinear_k(const float* __restrict__ in,
                                    int NB, int C, int H, int W,
                                    float* __restrict__ out, int OH, int OW,
                                    int Ctot, int c0, int add, int relu)
{
    size_t i = (size_t)blockIdx.x * blockDim.x + threadIdx.x;
    size_t total = (size_t)NB * C * OH * OW;
    if (i >= total) return;
    int ow = (int)(i % OW); size_t t = i / OW;
    int oh = (int)(t % OH); t /= OH;
    int c  = (int)(t % C);
    int b  = (int)(t / C);
    float ry = (OH > 1) ? (float)(H - 1) / (float)(OH - 1) : 0.f;
    float rx = (OW > 1) ? (float)(W - 1) / (float)(OW - 1) : 0.f;
    float fy = oh * ry, fx = ow * rx;
    int y0 = (int)floorf(fy), x0 = (int)floorf(fx);
    int y1 = min(y0 + 1, H - 1), x1 = min(x0 + 1, W - 1);
    float wy = fy - (float)y0, wx = fx - (float)x0;
    const float* base = in + ((size_t)b * C + c) * H * W;
    float v00 = base[(size_t)y0 * W + x0], v01 = base[(size_t)y0 * W + x1];
    float v10 = base[(size_t)y1 * W + x0], v11 = base[(size_t)y1 * W + x1];
    float v = (v00 * (1.f - wx) + v01 * wx) * (1.f - wy)
            + (v10 * (1.f - wx) + v11 * wx) * wy;
    size_t oidx = (((size_t)b * Ctot + (c0 + c)) * OH + oh) * OW + ow;
    if (add)  v += out[oidx];
    if (relu) v = fmaxf(v, 0.f);
    out[oidx] = v;
}

__global__ void relu_inplace_k(float* __restrict__ p, size_t n)
{
    size_t i = (size_t)blockIdx.x * blockDim.x + threadIdx.x;
    if (i < n) p[i] = fmaxf(p[i], 0.f);
}

// ---------------------------------------------------------------------------
// Host-side orchestration
// ---------------------------------------------------------------------------
namespace {

struct BN     { const float *g, *b, *m, *v; };
struct ConvBN { const _Float16* w; BN bn; };

struct Ctx {
    void* const* in; int i;
    hipStream_t st;
    char* base; size_t off;
    const float* t() { return (const float*)in[i++]; }
    BN bn() { BN r; r.g = t(); r.b = t(); r.m = t(); r.v = t(); return r; }
    float* falloc(size_t elems) {
        float* p = (float*)(base + off);
        off += ((elems * sizeof(float) + 255) & ~(size_t)255);
        return p;
    }
    _Float16* halloc(size_t elems) {
        _Float16* p = (_Float16*)(base + off);
        off += ((elems * sizeof(_Float16) + 255) & ~(size_t)255);
        return p;
    }
    ConvBN cb(int cout, int cin, int k) {
        const float* w = t();
        BN b = bn();
        size_t n = (size_t)cout * cin * k * k;
        _Float16* hw = halloc(n);
        unsigned blocks = (unsigned)((n + 255) / 256);
        f32_to_f16_k<<<blocks, 256, 0, st>>>(w, hw, n);
        ConvBN r; r.w = hw; r.bn = b; return r;
    }
};

struct Basic  { ConvBN c1, c2; };
struct Bottle { ConvBN c1, c2, c3, ds; bool has_ds; };
struct StageP { int n; int ch[4]; Basic br[4][4]; ConvBN fz[4][4]; };

struct Tensor { float* p; int B, C, H, W; };
static inline Tensor mk(float* p, int B, int C, int H, int W)
{ Tensor t; t.p = p; t.B = B; t.C = C; t.H = H; t.W = W; return t; }

static void convbn(hipStream_t st, const Tensor& x, const ConvBN& p, int K,
                   int stride, int pad, Tensor& o, const float* res,
                   int relu, int accum)
{
    int OH = (x.H + 2 * pad - K) / stride + 1;
    int OW = (x.W + 2 * pad - K) / stride + 1;
    o.B = x.B; o.H = OH; o.W = OW;
    dim3 grid((unsigned)((x.B * OH * OW + 63) / 64),
              (unsigned)((o.C + 63) / 64));
    bool ragged = ((x.C * K * K) & 31) != 0;
    if (K == 1)
        conv_bn_wmma<1, 1, 1, false><<<grid, 128, 0, st>>>(
            x.p, p.w, p.bn.g, p.bn.b, p.bn.m, p.bn.v, res, o.p,
            x.B, x.C, x.H, x.W, o.C, OH, OW, stride, relu, accum);
    else if (ragged)                      // stem conv (Cin=3, K=27)
        conv_bn_wmma<3, 3, 0, true><<<grid, 128, 0, st>>>(
            x.p, p.w, p.bn.g, p.bn.b, p.bn.m, p.bn.v, res, o.p,
            x.B, x.C, x.H, x.W, o.C, OH, OW, stride, relu, accum);
    else if (stride == 1)
        conv_bn_wmma<3, 3, 1, false><<<grid, 128, 0, st>>>(
            x.p, p.w, p.bn.g, p.bn.b, p.bn.m, p.bn.v, res, o.p,
            x.B, x.C, x.H, x.W, o.C, OH, OW, stride, relu, accum);
    else                                  // strided downsample convs (2/4/8)
        conv_bn_wmma<3, 3, 0, false><<<grid, 128, 0, st>>>(
            x.p, p.w, p.bn.g, p.bn.b, p.bn.m, p.bn.v, res, o.p,
            x.B, x.C, x.H, x.W, o.C, OH, OW, stride, relu, accum);
}

static void ups(hipStream_t st, const Tensor& in, float* outp, int OH, int OW,
                int Ctot, int c0, int add, int relu)
{
    size_t total = (size_t)in.B * in.C * OH * OW;
    unsigned blocks = (unsigned)((total + 255) / 256);
    upsample_bilinear_k<<<blocks, 256, 0, st>>>(
        in.p, in.B, in.C, in.H, in.W, outp, OH, OW, Ctot, c0, add, relu);
}

static void relu_ip(hipStream_t st, float* p, size_t n)
{
    unsigned blocks = (unsigned)((n + 255) / 256);
    relu_inplace_k<<<blocks, 256, 0, st>>>(p, n);
}

static StageP readStage(Ctx& c, int n)
{
    static const int CH[4] = {32, 64, 128, 256};
    StageP s; s.n = n;
    for (int i = 0; i < n; i++) {
        s.ch[i] = CH[i];
        for (int k = 0; k < 4; k++) {
            s.br[i][k].c1 = c.cb(CH[i], CH[i], 3);
            s.br[i][k].c2 = c.cb(CH[i], CH[i], 3);
        }
    }
    for (int j = 0; j < n; j++)
        for (int i = 0; i < n; i++)
            if (i != j)
                s.fz[j][i] = (i < j) ? c.cb(CH[j], CH[i], 3)
                                     : c.cb(CH[j], CH[i], 1);
    return s;
}

static void runStage(hipStream_t st, const StageP& sp, Tensor* xs, Tensor* ys,
                     float* TA, float* TB, float* TC)
{
    for (int i = 0; i < sp.n; i++) {
        Tensor X = xs[i];
        for (int k = 0; k < 4; k++) {
            float* op = (k == 3) ? xs[i].p : ((k & 1) ? TC : TB);
            Tensor t1 = mk(TA, X.B, X.C, X.H, X.W);
            convbn(st, X, sp.br[i][k].c1, 3, 1, 1, t1, nullptr, 1, 0);
            Tensor t2 = mk(op, X.B, X.C, X.H, X.W);
            convbn(st, t1, sp.br[i][k].c2, 3, 1, 1, t2, X.p, 1, 0);
            X = t2;
        }
    }
    for (int j = 0; j < sp.n; j++) {
        Tensor Y = ys[j];
        ups(st, xs[j], Y.p, Y.H, Y.W, Y.C, 0, 0, 0);          // identity init
        for (int i = 0; i < sp.n; i++) {
            if (i == j) continue;
            if (i < j) {                                      // strided 3x3 down
                Tensor o = Y;
                convbn(st, xs[i], sp.fz[j][i], 3, 1 << (j - i), 1,
                       o, nullptr, 0, /*accum=*/1);
            } else {                                          // 1x1 + upsample
                Tensor tmp = mk(TA, xs[i].B, Y.C, xs[i].H, xs[i].W);
                convbn(st, xs[i], sp.fz[j][i], 1, 1, 0, tmp, nullptr, 0, 0);
                ups(st, tmp, Y.p, Y.H, Y.W, Y.C, 0, /*add=*/1, 0);
            }
        }
        relu_ip(st, Y.p, (size_t)Y.B * Y.C * Y.H * Y.W);
    }
}

} // namespace

extern "C" void kernel_launch(void* const* d_in, const int* in_sizes, int n_in,
                              void* d_out, int out_size, void* d_ws, size_t ws_size,
                              hipStream_t stream)
{
    (void)in_sizes; (void)n_in; (void)out_size; (void)ws_size;
    hipStream_t st = stream;

    Ctx c{d_in, 0, st, (char*)d_ws, 0};

    // ---- parameters in setup_inputs() insertion order (convert weights) ----
    const float* ximg = c.t();
    ConvBN stc1 = c.cb(64, 3, 3);
    ConvBN stc2 = c.cb(64, 64, 3);
    Bottle L1[4];
    int cin = 64;
    for (int k = 0; k < 4; k++) {
        L1[k].c1 = c.cb(64, cin, 1);
        L1[k].c2 = c.cb(64, 64, 3);
        L1[k].c3 = c.cb(256, 64, 1);
        L1[k].has_ds = (k == 0);
        if (k == 0) L1[k].ds = c.cb(256, cin, 1);
        cin = 256;
    }
    ConvBN tr1a = c.cb(32, 256, 3);
    ConvBN tr1b = c.cb(64, 256, 3);
    StageP s2 = readStage(c, 2);
    ConvBN tr2 = c.cb(128, 64, 3);
    StageP s3 = readStage(c, 3);
    ConvBN tr3 = c.cb(256, 128, 3);
    StageP s4 = readStage(c, 4);

    // ---- activation slabs ----
    const size_t SLAB = (size_t)8 * 64 * 112 * 112;
    float* S[6];
    for (int i = 0; i < 6; i++) S[i] = c.falloc(SLAB);
    static const int chs[4] = {32, 64, 128, 256};
    static const int hh[4]  = {56, 28, 14, 7};
    float* XA[4]; float* XB[4];
    for (int i = 0; i < 4; i++) {
        size_t n = (size_t)8 * chs[i] * hh[i] * hh[i];
        XA[i] = c.falloc(n); XB[i] = c.falloc(n);
    }

    // ---- stem ----
    Tensor img = mk((float*)ximg, 8, 3, 224, 224);
    Tensor t0 = mk(S[0], 8, 64, 0, 0); convbn(st, img, stc1, 3, 2, 1, t0, nullptr, 1, 0);
    Tensor t1 = mk(S[1], 8, 64, 0, 0); convbn(st, t0,  stc2, 3, 2, 1, t1, nullptr, 1, 0);

    // ---- layer1: 4 bottlenecks @56x56 ----
    Tensor X = t1;
    static const int oslab[4] = {5, 1, 5, 1};
    for (int k = 0; k < 4; k++) {
        const Bottle& bp = L1[k];
        Tensor R;
        if (bp.has_ds) { R = mk(S[2], 8, 256, 0, 0); convbn(st, X, bp.ds, 1, 1, 0, R, nullptr, 0, 0); }
        else           { R = X; }
        Tensor T1 = mk(S[3], 8, 64, 0, 0);  convbn(st, X,  bp.c1, 1, 1, 0, T1, nullptr, 1, 0);
        Tensor T2 = mk(S[4], 8, 64, 0, 0);  convbn(st, T1, bp.c2, 3, 1, 1, T2, nullptr, 1, 0);
        Tensor O  = mk(S[oslab[k]], 8, 256, 0, 0);
        convbn(st, T2, bp.c3, 1, 1, 0, O, R.p, 1, 0);
        X = O;
    }

    // ---- transition 1 ----
    Tensor setA[4], setB[4];
    for (int i = 0; i < 4; i++) {
        setA[i] = mk(XA[i], 8, chs[i], hh[i], hh[i]);
        setB[i] = mk(XB[i], 8, chs[i], hh[i], hh[i]);
    }
    convbn(st, X, tr1a, 3, 1, 1, setA[0], nullptr, 1, 0);
    convbn(st, X, tr1b, 3, 2, 1, setA[1], nullptr, 1, 0);

    Tensor* xs = setA; Tensor* ys = setB;

    runStage(st, s2, xs, ys, S[0], S[2], S[3]);
    { Tensor* t = xs; xs = ys; ys = t; }
    convbn(st, xs[1], tr2, 3, 2, 1, xs[2], nullptr, 1, 0);

    runStage(st, s3, xs, ys, S[0], S[2], S[3]);
    { Tensor* t = xs; xs = ys; ys = t; }
    convbn(st, xs[2], tr3, 3, 2, 1, xs[3], nullptr, 1, 0);

    runStage(st, s4, xs, ys, S[0], S[2], S[3]);
    { Tensor* t = xs; xs = ys; ys = t; }

    // ---- outputs ----
    float* feat = (float*)d_out;
    const size_t featN = (size_t)8 * 480 * 56 * 56;
    const size_t h0N   = (size_t)8 * 32 * 56 * 56;
    ups(st, xs[0], feat, 56, 56, 480,   0, 0, 0);
    ups(st, xs[1], feat, 56, 56, 480,  32, 0, 0);
    ups(st, xs[2], feat, 56, 56, 480,  96, 0, 0);
    ups(st, xs[3], feat, 56, 56, 480, 224, 0, 0);
    ups(st, xs[0], feat + featN,        56, 56,  32, 0, 0, 0);
    ups(st, xs[3], feat + featN + h0N,  56, 56, 256, 0, 0, 0);
}
